// Generator_79310866088539
// MI455X (gfx1250) — matrix operand order repaired and checked by reading, not compiled, not run on hardware
//
#include <hip/hip_runtime.h>

// ---------------- problem constants (match reference) ----------------
constexpr int Bsz  = 2;
constexpr int F    = 24576;
constexpr int CI   = 256;
constexpr int CO   = 256;
constexpr int K    = 9;
constexpr int WD   = 512;
constexpr int Ntot = Bsz * F;       // 49152
constexpr int KK   = K * CI;        // 2304 (reduction dim of the big GEMM)

typedef __bf16 bf16_t;
typedef __attribute__((ext_vector_type(16))) __bf16 v16bf;
typedef __attribute__((ext_vector_type(8)))  __bf16 v8bf;
typedef __attribute__((ext_vector_type(8)))  float  v8f;

// ---------------- prep kernel 1: styles = w @ (A^T / sqrt(WD)) + bias ----
__global__ void styles_kernel(const float* __restrict__ w,
                              const float* __restrict__ aw,   // [CI, WD]
                              const float* __restrict__ ab,   // [CI]
                              float* __restrict__ styles)     // [Bsz, CI]
{
    const int b = blockIdx.x;
    const int i = threadIdx.x;          // 0..CI-1
    const float wg = 0.04419417382415922f;   // 1/sqrt(512)
    float s = 0.f;
    for (int j = 0; j < WD; ++j)
        s += w[b * WD + j] * aw[i * WD + j];
    styles[b * CI + i] = s * wg + ab[i];
}

// ---------------- prep kernel 2: demod d[b,o] = rsqrt(sum wmod^2 + eps) --
__global__ void demod_kernel(const float* __restrict__ cw,     // [CO, CI, 1, K]
                             const float* __restrict__ styles, // [Bsz, CI]
                             float* __restrict__ dem)          // [Bsz, CO]
{
    const int o    = blockIdx.x;
    const int b    = blockIdx.y;
    const int lane = threadIdx.x;       // 32 threads = 1 wave
    float s = 0.f;
    for (int r = lane; r < KK; r += 32) {
        const int i = r / K;            // r = i*K + k
        const float wv = cw[o * KK + r] * styles[b * CI + i];
        s += wv * wv;
    }
    for (int off = 16; off > 0; off >>= 1)
        s += __shfl_down(s, off, 32);
    if (lane == 0)
        dem[b * CO + o] = rsqrtf(s + 1e-8f);
}

// ---------------- prep kernel 3: WbT[b][o][kn*CI+i] (bf16, K-major rows) --
__global__ void buildw_kernel(const float* __restrict__ cw,     // [CO, CI, 1, K]
                              const float* __restrict__ styles, // [Bsz, CI]
                              const float* __restrict__ dem,    // [Bsz, CO]
                              bf16_t* __restrict__ wbt)         // [Bsz, CO, KK]
{
    const int o = blockIdx.x;
    const int b = blockIdx.y;
    const float dv = dem[b * CO + o];
    bf16_t* dst = wbt + ((size_t)b * CO + o) * KK;
    for (int r = threadIdx.x; r < KK; r += blockDim.x) {
        const int i  = r % CI;          // r = kn*CI + i
        const int kn = r / CI;
        const float v = cw[o * KK + i * K + kn] * styles[b * CI + i] * dv;
        dst[r] = (bf16_t)v;
    }
}

// ---------------- prep kernel 4: x (f32) -> xb (bf16), + zero row --------
__global__ void xcvt_kernel(const float* __restrict__ x,  // [Ntot*CI]
                            bf16_t* __restrict__ xb,      // [Ntot*CI]
                            bf16_t* __restrict__ zrow)    // [CI] zeros
{
    const int idx = blockIdx.x * blockDim.x + threadIdx.x;  // 8 floats each
    const float4* px = (const float4*)x + (size_t)idx * 2;
    const float4 a = px[0], b = px[1];
    const float fa[8] = { a.x, a.y, a.z, a.w, b.x, b.y, b.z, b.w };
    v8bf v;
#pragma unroll
    for (int e = 0; e < 8; ++e) v[e] = (bf16_t)fa[e];
    *((v8bf*)xb + idx) = v;

    if (blockIdx.x == 0 && threadIdx.x < CI / 8) {   // zero 512 bytes once
        const v8bf z = {};
        *((v8bf*)zrow + threadIdx.x) = z;
    }
}

// ---------------- main kernel: gathered GEMM on the WMMA pipe ------------
// Block tile: 256 faces x 128 out-channels; 8 waves as 4 (M) x 2 (N);
// each wave: 4 M-tiles x 4 N-tiles of 16x16 (64x64), K-step 32 bf16 WMMA.
// Per K-step: 8 A-loads + 8 B-loads (b128) feed 16 WMMAs -> 1.0 load/WMMA.
#define TMB 256
#define TNB 128

__global__ __launch_bounds__(256)
void faceconv_wmma(const bf16_t* __restrict__ xb,          // [Ntot, CI] bf16
                   const bf16_t* __restrict__ zrow,        // [CI] zeros
                   const int* __restrict__ nbr,            // [Ntot, K]
                   const unsigned char* __restrict__ ispad,// [Ntot, K] (bool)
                   const bf16_t* __restrict__ wbt,         // [Bsz, CO, KK]
                   const float* __restrict__ noise_const,  // [F]
                   const float* __restrict__ noise_strength,// [1]
                   const float* __restrict__ bias,         // [CO]
                   float* __restrict__ out)                // [Ntot, CO]
{
    const int b     = blockIdx.z;
    const int m0    = blockIdx.x * TMB;    // face base within batch
    const int n0    = blockIdx.y * TNB;    // out-channel base
    const int tid   = threadIdx.x;
    const int lane  = tid & 31;
    const int wave  = tid >> 5;            // 0..7
    const int wm    = wave & 3;            // M direction: 64 rows each
    const int wn    = wave >> 2;           // N direction: 64 cols each
    const int lrow  = lane & 15;
    const int khalf = lane >> 4;           // which K-half this lane holds

    const bf16_t* wb = wbt + (size_t)b * CO * KK;

    const v8f vzero = {};
    v8f acc[4][4];
#pragma unroll
    for (int t = 0; t < 4; ++t)
#pragma unroll
        for (int j = 0; j < 4; ++j) acc[t][j] = vzero;

    // A-fragment face rows (M = lane%16 for both lane halves, per ISA layout)
    int frow[4];
#pragma unroll
    for (int t = 0; t < 4; ++t)
        frow[t] = m0 + wm * 64 + t * 16 + lrow;

    // B row bases for this lane's 4 N-tiles (column N = lane%16)
    const bf16_t* brow[4];
#pragma unroll
    for (int j = 0; j < 4; ++j)
        brow[j] = wb + (size_t)(n0 + wn * 64 + j * 16 + lrow) * KK;

    union AFrag { uint4 u[2]; v16bf v; };

#pragma unroll 1
    for (int kn = 0; kn < K; ++kn) {
        // gather row pointers for this neighbor (padded/OOB -> zero row)
        const bf16_t* src[4];
#pragma unroll
        for (int t = 0; t < 4; ++t) {
            const int gf  = b * F + frow[t];
            const int idx = nbr[gf * K + kn];
            const bool pad = (ispad[gf * K + kn] != 0) ||
                             ((unsigned)idx >= (unsigned)Ntot);
            src[t] = pad ? zrow : (xb + (size_t)idx * CI);
        }

#pragma unroll 1
        for (int c0 = 0; c0 < CI; c0 += 32) {
            // ---- A fragments: two 16B loads straight into WMMA layout ----
            // lanes 0-15 hold K[0..15]; lanes 16-31 K[16..31]
            v16bf afrag[4];
#pragma unroll
            for (int t = 0; t < 4; ++t) {
                AFrag a;
                a.u[0] = *(const uint4*)(src[t] + c0 + khalf * 16);
                a.u[1] = *(const uint4*)(src[t] + c0 + khalf * 16 + 8);
                afrag[t] = a.v;
            }

            // ---- B fragments: 32 contiguous bytes per lane from WbT ------
            // lane holds column N = lane%16, K rows khalf*16..+15 (packed)
            const int rbase = kn * CI + c0 + khalf * 16;
            v16bf bfrag[4];
#pragma unroll
            for (int j = 0; j < 4; ++j)
                bfrag[j] = *(const v16bf*)(brow[j] + rbase);

            // ---- 16 WMMAs per K-step -------------------------------------
#pragma unroll
            for (int t = 0; t < 4; ++t)
#pragma unroll
                for (int j = 0; j < 4; ++j)
                    acc[t][j] = __builtin_amdgcn_wmma_f32_16x16x32_bf16(
                        false, afrag[t], false, bfrag[j],
                        (short)0, acc[t][j], false, false);
        }
    }

    // ---- epilogue: noise + bias + leaky_relu*sqrt(2) + clamp --------------
    const float nstr = noise_strength[0];
    float bv[4];
#pragma unroll
    for (int j = 0; j < 4; ++j)
        bv[j] = bias[n0 + wn * 64 + j * 16 + lrow];

#pragma unroll
    for (int t = 0; t < 4; ++t) {
#pragma unroll
        for (int e = 0; e < 8; ++e) {
            const int fr = m0 + wm * 64 + t * 16 + khalf * 8 + e; // row in batch
            const float nv = noise_const[fr] * nstr;
            float* orow = out + ((size_t)(b * F + fr)) * CO;
#pragma unroll
            for (int j = 0; j < 4; ++j) {
                const int oc = n0 + wn * 64 + j * 16 + lrow;
                float v = acc[t][j][e];
                v += nv;
                v += bv[j];
                v = (v > 0.f ? v : 0.2f * v) * 1.4142135623730951f;
                v = fminf(fmaxf(v, -256.f), 256.f);
                orow[oc] = v;
            }
        }
    }
}

// ---------------- launcher ----------------------------------------------
extern "C" void kernel_launch(void* const* d_in, const int* in_sizes, int n_in,
                              void* d_out, int out_size, void* d_ws, size_t ws_size,
                              hipStream_t stream) {
    const float*         x   = (const float*)d_in[0];
    const float*         w   = (const float*)d_in[1];
    const int*           nbr = (const int*)d_in[2];
    const unsigned char* pad = (const unsigned char*)d_in[3]; // bool mask
    // d_in[4] = pad_size scalar (unused; indices are clamped/masked)
    const float*         aw  = (const float*)d_in[5];
    const float*         ab  = (const float*)d_in[6];
    const float*         cw  = (const float*)d_in[7];
    const float*         nc  = (const float*)d_in[8];
    const float*         ns  = (const float*)d_in[9];
    const float*         bia = (const float*)d_in[10];
    float*               out = (float*)d_out;

    char* ws = (char*)d_ws;
    float*  styles = (float*)ws;                        // 2*256 f32
    float*  dem    = (float*)(ws + 4096);               // 2*256 f32
    bf16_t* wbt    = (bf16_t*)(ws + 8192);              // 2*256*2304 bf16 (2.36 MB)
    bf16_t* zrow   = (bf16_t*)(ws + 2367488);           // 256 bf16 zeros (512 B)
    bf16_t* xbf    = (bf16_t*)(ws + 2368000);           // Ntot*CI bf16 (25.2 MB)

    styles_kernel<<<dim3(Bsz), dim3(CI), 0, stream>>>(w, aw, ab, styles);
    demod_kernel <<<dim3(CO, Bsz), dim3(32), 0, stream>>>(cw, styles, dem);
    buildw_kernel<<<dim3(CO, Bsz), dim3(256), 0, stream>>>(cw, styles, dem, wbt);
    xcvt_kernel  <<<dim3((Ntot * CI / 8) / 256), dim3(256), 0, stream>>>(x, xbf, zrow);

    faceconv_wmma<<<dim3(F / TMB, CO / TNB, Bsz), dim3(256), 0, stream>>>(
        xbf, zrow, nbr, pad, wbt, nc, ns, bia, out);
}
